// Attention_17867063951949
// MI455X (gfx1250) — compile-verified, matrix operand order checked by reading, not run on hardware
//
#include <hip/hip_runtime.h>
#include <hip/hip_bf16.h>

// ---------------- problem constants (match reference) ----------------
#define BATCH 16
#define QL    4
#define NTOK  (BATCH * QL)   // 64
#define KVLEN 4096
#define NH    32             // H
#define NKVH  8              // KVH
#define GQA   (NH / NKVH)    // 4
#define HD    128            // D
#define DIM   4096
#define THETA 10000.0f

#define LDS_KV_STRIDE 132    // 128 + 4 pad (matches TDM pad_interval=128dw, pad=4dw)

typedef __attribute__((ext_vector_type(16))) __bf16 v16bf;
typedef __attribute__((ext_vector_type(8)))  float  v8f;
typedef unsigned int __attribute__((ext_vector_type(4))) u32x4;
typedef int          __attribute__((ext_vector_type(4))) i32x4;
typedef int          __attribute__((ext_vector_type(8))) i32x8;

#if __has_builtin(__builtin_amdgcn_tensor_load_to_lds)
#define HAVE_TDM 1
#else
#define HAVE_TDM 0
#endif

// WMMA 16-bit A/B fragment K-index for element e of a lane (ISA 7.12.2)
__device__ __forceinline__ int wmma_kidx(int e, int half8) {
    return (e & 7) + ((e & 8) << 1) + half8;   // half8 = 8 if lane>=16 else 0
}

__device__ __forceinline__ v8f wmma_bf16(v16bf a, v16bf b, v8f c) {
    return __builtin_amdgcn_wmma_f32_16x16x32_bf16(false, a, false, b,
                                                   (short)0, c, false, false);
}

// -------------------------------------------------------------------
// TDM: DMA one 32-row x 128-float tile (row stride 1024 floats in global)
// into LDS with +4 DWORD padding per 128 DWORDs (row stride 132 floats).
// Descriptor bit layout per CDNA5 ISA 8.3/8.4.
// -------------------------------------------------------------------
__device__ __forceinline__ void tdm_load_row_tile(const float* gsrc, float* ldst)
{
#if HAVE_TDM
    unsigned long long ga = (unsigned long long)(uintptr_t)gsrc;
    // flat shared addr = {shared aperture, lds offset} -> low 32 bits = LDS addr
    unsigned int la = (unsigned int)(uintptr_t)ldst;

    u32x4 g0;
    g0.x = 1u;                                                  // count=1 (valid)
    g0.y = la;                                                  // lds_addr
    g0.z = (unsigned int)ga;                                    // global_addr[31:0]
    g0.w = ((unsigned int)(ga >> 32) & 0x01FFFFFFu) | (2u << 30); // addr[56:32], type=2

    i32x8 g1;
    // data_size=2(4B), pad_enable, pad_interval=6(128dw), pad_amount=3(4dw)
    g1[0] = (int)((2u << 16) | (1u << 20) | (6u << 22) | (3u << 25));
    g1[1] = (int)(128u << 16);                 // tensor_dim0[15:0]=128 (bits 63:48)
    g1[2] = (int)(32u << 16);                  // tensor_dim0 hi=0 | tensor_dim1=32
    g1[3] = (int)(128u << 16);                 // tensor_dim1 hi=0 | tile_dim0=128
    g1[4] = 32;                                // tile_dim1=32, tile_dim2=0
    g1[5] = 1024;                              // tensor_dim0_stride = NKVH*HD
    g1[6] = 0;                                 // stride hi / dim1_stride lo
    g1[7] = 0;

    i32x4 z4 = {0, 0, 0, 0};
#if defined(__clang_major__) && (__clang_major__ >= 23)
    i32x8 z8 = {0, 0, 0, 0, 0, 0, 0, 0};
    __builtin_amdgcn_tensor_load_to_lds(g0, g1, z4, z4, z8, 0);
#else
    __builtin_amdgcn_tensor_load_to_lds(g0, g1, z4, z4, 0);
#endif
#endif
}

__device__ __forceinline__ void tdm_wait()
{
#if HAVE_TDM && __has_builtin(__builtin_amdgcn_s_wait_tensorcnt)
    __builtin_amdgcn_s_wait_tensorcnt(0);
#endif
    asm volatile("" ::: "memory");   // keep LDS reads below the wait
}

// Coalesced fallback staging (boundary tiles with fresh K/V rows, or no TDM).
__device__ __forceinline__ void stage_tile_manual(float (*dst)[LDS_KV_STRIDE],
                                                  const float* __restrict__ cache,
                                                  const float* __restrict__ fresh,
                                                  int b, int t0, int kvh, int clen,
                                                  int lane)
{
    for (int r = 0; r < 32; ++r) {
        int t = t0 + r;
        float4 val = make_float4(0.f, 0.f, 0.f, 0.f);
        if (t < clen) {
            const float* src = cache + ((((size_t)b * KVLEN) + t) * NKVH + kvh) * HD;
            val = *(const float4*)(src + lane * 4);
        } else if (t < clen + QL) {
            const float* src =
                fresh + (((size_t)(b * QL + (t - clen))) * NKVH + kvh) * HD;
            val = *(const float4*)(src + lane * 4);
        }
        *(float4*)(&dst[r][lane * 4]) = val;
    }
}

// -------------------------------------------------------------------
// GEMM: C[M,N] = A[M,K] @ W[K,N] (both row major f32), bf16 WMMA, f32 acc.
// One wave computes a 16x64 strip (4 N-tiles): the A fragment is reused 4x
// and the 32x64 W tile is staged through LDS with coalesced float4 loads
// (one bulk wait per tile instead of 64 serialized b32 load-use chains),
// then consumed via bank-conflict-free DS reads.
// -------------------------------------------------------------------
__global__ __launch_bounds__(32)
void gemm_rowmajor_bf16(const float* __restrict__ A, const float* __restrict__ W,
                        float* __restrict__ C, int M, int N, int K)
{
    __shared__ float lds_w[32][68];              // 32 k-rows x 64 cols, +4 pad

    const int lane  = threadIdx.x & 31;
    const int lmod  = lane & 15;
    const int half8 = (lane & 16) >> 1;
    const int n0 = blockIdx.x * 64;
    const int m0 = blockIdx.y * 16;

    const int srow = lane >> 4;                  // staging: row parity
    const int scol = lane & 15;                  // staging: float4 column

    v8f acc[4] = {{}, {}, {}, {}};
    const float* Arow = A + (size_t)(m0 + lmod) * K;

    for (int k0 = 0; k0 < K; k0 += 32) {
        // ---- coalesced stage of W tile [32 x 64] into LDS ----
        const float* Wt = W + (size_t)k0 * N + n0;
        __builtin_prefetch(Wt + (size_t)32 * N, 0, 3);      // next W tile
#pragma unroll
        for (int it = 0; it < 16; ++it) {
            int row = it * 2 + srow;
            float4 v4 = *(const float4*)(Wt + (size_t)row * N + scol * 4);
            *(float4*)(&lds_w[row][scol * 4]) = v4;
        }
        __syncthreads();

        const float* ap = Arow + k0 + half8;
        __builtin_prefetch(ap + 32, 0, 3);                  // next A strip
        v16bf a;
#pragma unroll
        for (int e = 0; e < 8; ++e) {
            a[e]     = (__bf16)ap[e];
            a[e + 8] = (__bf16)ap[e + 16];
        }

#pragma unroll
        for (int t = 0; t < 4; ++t) {
            v16bf bf;
#pragma unroll
            for (int e = 0; e < 16; ++e)
                bf[e] = (__bf16)lds_w[wmma_kidx(e, half8)][t * 16 + lmod];
            acc[t] = wmma_bf16(a, bf, acc[t]);
        }
        __syncthreads();                                    // reuse lds_w safely
    }
#pragma unroll
    for (int t = 0; t < 4; ++t)
#pragma unroll
        for (int v = 0; v < 8; ++v)
            C[(size_t)(m0 + v + half8) * N + n0 + t * 16 + lmod] = acc[t][v];
}

// -------------------------------------------------------------------
// Interleaved RoPE on Q [NTOK, NH, HD] and K_new [NTOK, NKVH, HD]
// -------------------------------------------------------------------
__global__ __launch_bounds__(256)
void rope_kernel(float* __restrict__ q, float* __restrict__ k,
                 const int* __restrict__ cache_len)
{
    const int n = blockIdx.x;
    const int b = n / QL, l = n % QL;
    const float pos = (float)(cache_len[b] + l);
    const int RHALF = HD / 2;
    const int TOTAL = (NH + NKVH) * RHALF;

    for (int i = threadIdx.x; i < TOTAL; i += blockDim.x) {
        int head = i / RHALF;
        int j    = i % RHALF;
        float inv = __powf(THETA, -(float)j / (float)RHALF);
        float s, c;
        __sincosf(pos * inv, &s, &c);
        float* base = (head < NH)
            ? (q + ((size_t)n * NH   + head)        * HD)
            : (k + ((size_t)n * NKVH + (head - NH)) * HD);
        float te = base[2 * j], to = base[2 * j + 1];
        base[2 * j]     = te * c - to * s;
        base[2 * j + 1] = to * c + te * s;
    }
}

// -------------------------------------------------------------------
// Flash attention, one wave per (b, kvh). 16 q-rows = G*QL = one WMMA tile.
// K/V tiles staged in LDS via TDM (pure-cache tiles) or coalesced copies.
// -------------------------------------------------------------------
__global__ __launch_bounds__(32)
void attn_kernel(const float* __restrict__ q,  const float* __restrict__ kn,
                 const float* __restrict__ vn, const float* __restrict__ ck,
                 const float* __restrict__ cv, const int* __restrict__ cache_len,
                 float* __restrict__ out)
{
    __shared__ struct {
        float k[32][LDS_KV_STRIDE];
        float v[32][LDS_KV_STRIDE];
        float p[16][36];
    } smem;

    const int kvh   = blockIdx.x;
    const int b     = blockIdx.y;
    const int lane  = threadIdx.x & 31;
    const int lmod  = lane & 15;
    const int half8 = (lane & 16) >> 1;
    const int clen  = cache_len[b];
    const float scale = 0.08838834764831845f;     // 1/sqrt(128)

    // Preload Q as 4 A-fragments (row r = g*QL + l -> lane lmod)
    v16bf qa[4];
    {
        int g = lmod >> 2, l = lmod & 3;
        const float* qrow =
            q + (((size_t)(b * QL + l)) * NH + (kvh * GQA + g)) * HD;
#pragma unroll
        for (int c2 = 0; c2 < 4; ++c2)
#pragma unroll
            for (int e = 0; e < 16; ++e)
                qa[c2][e] = (__bf16)qrow[c2 * 32 + wmma_kidx(e, half8)];
    }

    v8f o[8];
#pragma unroll
    for (int j = 0; j < 8; ++j) o[j] = (v8f){};
    float m_arr[8], l_arr[8];
#pragma unroll
    for (int v = 0; v < 8; ++v) { m_arr[v] = -1e30f; l_arr[v] = 0.0f; }

    const int tmax = clen + QL;
    for (int t0 = 0; t0 < tmax; t0 += 32) {
        const bool pure = HAVE_TDM && (t0 + 32 <= clen);
        if (pure) {
            const float* kg = ck + ((((size_t)b * KVLEN) + t0) * NKVH + kvh) * HD;
            const float* vg = cv + ((((size_t)b * KVLEN) + t0) * NKVH + kvh) * HD;
            tdm_load_row_tile(kg, &smem.k[0][0]);
            tdm_load_row_tile(vg, &smem.v[0][0]);
            __builtin_prefetch(kg + 32 * NKVH * HD, 0, 2);   // next K tile
            tdm_wait();
        } else {
            stage_tile_manual(smem.k, ck, kn, b, t0, kvh, clen, lane);
            stage_tile_manual(smem.v, cv, vn, b, t0, kvh, clen, lane);
        }
        __syncthreads();

        // ---- S = Q @ K^T : two 16x16 tiles over the 32 staged keys ----
        v8f s0 = {}, s1 = {};
        {
            const float* kr0 = &smem.k[lmod][0];
            const float* kr1 = &smem.k[lmod + 16][0];
#pragma unroll
            for (int c2 = 0; c2 < 4; ++c2) {
                v16bf kb0, kb1;
#pragma unroll
                for (int e = 0; e < 16; ++e) {
                    int d = c2 * 32 + wmma_kidx(e, half8);
                    kb0[e] = (__bf16)kr0[d];
                    kb1[e] = (__bf16)kr1[d];
                }
                s0 = wmma_bf16(qa[c2], kb0, s0);
                s1 = wmma_bf16(qa[c2], kb1, s1);
            }
        }

        // ---- scale + causal mask + online softmax ----
        const int key0 = t0 + lmod, key1 = t0 + 16 + lmod;
#pragma unroll
        for (int v = 0; v < 8; ++v) {
            int m   = v + half8;                  // q-row 0..15
            int pos = clen + (m & 3);             // l = m % QL
            float x0 = s0[v] * scale;
            float x1 = s1[v] * scale;
            if (key0 > pos) x0 = -1e30f;
            if (key1 > pos) x1 = -1e30f;

            float mx = fmaxf(x0, x1);
            mx = fmaxf(mx, __shfl_xor(mx, 1));
            mx = fmaxf(mx, __shfl_xor(mx, 2));
            mx = fmaxf(mx, __shfl_xor(mx, 4));
            mx = fmaxf(mx, __shfl_xor(mx, 8));

            float mnew  = fmaxf(m_arr[v], mx);
            float alpha = __expf(m_arr[v] - mnew);
            float p0 = __expf(x0 - mnew);
            float p1 = __expf(x1 - mnew);
            m_arr[v] = mnew;

            float sum = p0 + p1;
            sum += __shfl_xor(sum, 1);
            sum += __shfl_xor(sum, 2);
            sum += __shfl_xor(sum, 4);
            sum += __shfl_xor(sum, 8);
            l_arr[v] = l_arr[v] * alpha + sum;

#pragma unroll
            for (int j = 0; j < 8; ++j) o[j][v] = o[j][v] * alpha;

            smem.p[m][lmod]      = p0;            // C-layout -> LDS bounce
            smem.p[m][16 + lmod] = p1;
        }
        __syncthreads();

        // ---- re-layout P as A-fragment (DS reads perform the transpose) ----
        v16bf pa;
#pragma unroll
        for (int e = 0; e < 16; ++e)
            pa[e] = (__bf16)smem.p[lmod][wmma_kidx(e, half8)];

        // ---- O += P @ V : 8 N-tiles over D=128, V from padded LDS ----
#pragma unroll
        for (int j = 0; j < 8; ++j) {
            v16bf vb;
#pragma unroll
            for (int e = 0; e < 16; ++e)
                vb[e] = (__bf16)smem.v[wmma_kidx(e, half8)][j * 16 + lmod];
            o[j] = wmma_bf16(pa, vb, o[j]);
        }
        __syncthreads();   // protect smem before next tile is staged
    }

    // ---- normalize and scatter to [n, h, d] ----
#pragma unroll
    for (int v = 0; v < 8; ++v) {
        int m = v + half8;
        int g = m >> 2, l = m & 3;
        float inv = 1.0f / l_arr[v];
        float* orow =
            out + (((size_t)(b * QL + l)) * NH + (kvh * GQA + g)) * HD;
#pragma unroll
        for (int j = 0; j < 8; ++j)
            orow[j * 16 + lmod] = o[j][v] * inv;
    }
}

// -------------------------------------------------------------------
extern "C" void kernel_launch(void* const* d_in, const int* in_sizes, int n_in,
                              void* d_out, int out_size, void* d_ws, size_t ws_size,
                              hipStream_t stream)
{
    const float* x  = (const float*)d_in[0];
    const float* wq = (const float*)d_in[1];
    const float* wk = (const float*)d_in[2];
    const float* wv = (const float*)d_in[3];
    const float* wo = (const float*)d_in[4];
    const float* ck = (const float*)d_in[5];
    const float* cv = (const float*)d_in[6];
    const int*  cl  = (const int*)d_in[7];

    float* q    = (float*)d_ws;                       // [64, 4096]
    float* kn   = q  + (size_t)NTOK * NH   * HD;      // [64, 1024]
    float* vn   = kn + (size_t)NTOK * NKVH * HD;      // [64, 1024]
    float* attn = vn + (size_t)NTOK * NKVH * HD;      // [64, 4096]

    dim3 wave(32, 1, 1);
    gemm_rowmajor_bf16<<<dim3((NH * HD) / 64, NTOK / 16), wave, 0, stream>>>(
        x, wq, q, NTOK, NH * HD, DIM);
    gemm_rowmajor_bf16<<<dim3((NKVH * HD) / 64, NTOK / 16), wave, 0, stream>>>(
        x, wk, kn, NTOK, NKVH * HD, DIM);
    gemm_rowmajor_bf16<<<dim3((NKVH * HD) / 64, NTOK / 16), wave, 0, stream>>>(
        x, wv, vn, NTOK, NKVH * HD, DIM);

    rope_kernel<<<NTOK, 256, 0, stream>>>(q, kn, cl);

    attn_kernel<<<dim3(NKVH, BATCH), wave, 0, stream>>>(
        q, kn, vn, ck, cv, cl, attn);

    gemm_rowmajor_bf16<<<dim3(DIM / 64, NTOK / 16), wave, 0, stream>>>(
        attn, wo, (float*)d_out, NTOK, DIM, NH * HD);
}
